// encoder_3350074491391
// MI455X (gfx1250) — compile-verified, matrix operand order checked by reading, not run on hardware
//
#include <hip/hip_runtime.h>
#include <cstddef>

// ---------------------------------------------------------------------------
// Point-transformer encoder for MI455X (gfx1250, wave32, WMMA).
// All matmul-shaped work uses v_wmma_f32_16x16x32_bf16 (f32 accumulate).
// Per-point attention is fully fused (gather + pos-MLP + attn-MLP + softmax +
// weighted sum) with the [32 x d] A-matrices staged in LDS.
// ---------------------------------------------------------------------------

typedef __attribute__((ext_vector_type(16))) __bf16 v16bf;
typedef __attribute__((ext_vector_type(8)))  __bf16 v8bf;
typedef __attribute__((ext_vector_type(8)))  float  v8f;

__device__ inline __bf16 f2bf(float f) {
    unsigned u = __builtin_bit_cast(unsigned, f);
    unsigned r = (u + 0x7FFFu + ((u >> 16) & 1u)) >> 16;   // RNE
    unsigned short s = (unsigned short)r;
    return __builtin_bit_cast(__bf16, s);
}

__device__ inline v8f vzero() {
    v8f z;
#pragma unroll
    for (int i = 0; i < 8; ++i) z[i] = 0.f;
    return z;
}

__device__ inline v8f wmma_bf16(v16bf a, v16bf b, v8f c) {
    return __builtin_amdgcn_wmma_f32_16x16x32_bf16(
        false, a, false, b, (short)0, c, false, false);
}

// Load one 16x32 bf16 tile (A layout; also B layout when base = W^T row-major).
// lane l<16 : row l, halves k0+0..7  and k0+16..23
// lane l+16 : row l, halves k0+8..15 and k0+24..31
__device__ inline v16bf ld_tile(const __bf16* base, int ld, int row0, int k0, int lane) {
    int l  = lane & 15;
    int hi = (lane >> 4) & 1;
    const __bf16* p = base + (size_t)(row0 + l) * ld + k0 + hi * 8;
    v8bf lo = *(const v8bf*)p;
    v8bf hh = *(const v8bf*)(p + 16);
    v16bf r;
#pragma unroll
    for (int i = 0; i < 8; ++i) { r[i] = lo[i]; r[i + 8] = hh[i]; }
    return r;
}

// ---------------------------------------------------------------------------
// KNN: one thread per point, stable top-24 on (ordered_float<<32 | index).
// ---------------------------------------------------------------------------
__global__ __launch_bounds__(64) void knn_kernel(const float* __restrict__ xyz,
                                                 int* __restrict__ knn) {
    int t = blockIdx.x * 64 + threadIdx.x;           // 0..16383
    int bb = t >> 12;
    const float* base = xyz + (size_t)bb * 4096 * 3;
    const float* q = xyz + (size_t)t * 3;
    float qx = q[0], qy = q[1], qz = q[2];
    float sn = qx * qx + qy * qy + qz * qz;
    unsigned long long best[24];
#pragma unroll
    for (int i = 0; i < 24; ++i) best[i] = ~0ULL;
    for (int m = 0; m < 4096; ++m) {
        float x = base[m * 3 + 0], y = base[m * 3 + 1], z = base[m * 3 + 2];
        float sm  = x * x + y * y + z * z;
        float dot = qx * x + qy * y + qz * z;
        float d = sn + sm - 2.0f * dot;              // same formula as reference
        unsigned u  = __builtin_bit_cast(unsigned, d);
        unsigned ou = (u & 0x80000000u) ? ~u : (u | 0x80000000u);
        unsigned long long key = ((unsigned long long)ou << 32) | (unsigned)m;
        if (key < best[23]) {
            int p = 23;
            while (p > 0 && best[p - 1] > key) { best[p] = best[p - 1]; --p; }
            best[p] = key;
        }
    }
    for (int i = 0; i < 24; ++i) knn[(size_t)t * 24 + i] = (int)(best[i] & 0xFFFFFFFFu);
}

// ---------------------------------------------------------------------------
// Layer-1 fc1: [M,3] @ [3,32] + b  (too thin for WMMA) -> bf16
// ---------------------------------------------------------------------------
__global__ __launch_bounds__(256) void fc1_l1_kernel(const float* __restrict__ xyz,
                                                     const float* __restrict__ W,
                                                     const float* __restrict__ b,
                                                     __bf16* __restrict__ xb, int M) {
    int t = blockIdx.x * 256 + threadIdx.x;
    if (t >= M * 32) return;
    int r = t >> 5, c = t & 31;
    const float* p = xyz + (size_t)r * 3;
    float v = p[0] * W[c] + p[1] * W[32 + c] + p[2] * W[64 + c] + b[c];
    xb[t] = f2bf(v);
}

// W [Kd][Nd] f32 -> Wt [Nd][Kd] bf16
__global__ __launch_bounds__(256) void transpose_kernel(const float* __restrict__ W,
                                                        __bf16* __restrict__ Wt,
                                                        int Kd, int Nd) {
    int t = blockIdx.x * 256 + threadIdx.x;
    if (t >= Kd * Nd) return;
    int k = t / Nd, n = t - k * Nd;
    Wt[(size_t)n * Kd + k] = f2bf(W[t]);
}

// ---------------------------------------------------------------------------
// Dense GEMM: out = A[M,Kd](bf16) @ Bt[Nd,Kd]^T(bf16) + bias, f32/bf16 out.
// Block = 128 threads = 4 waves; each wave computes a 16 x (16*NT) tile.
// ---------------------------------------------------------------------------
template <int NT>
__global__ __launch_bounds__(128) void gemm_kernel(const __bf16* __restrict__ A,
                                                   const __bf16* __restrict__ Bt,
                                                   const float* __restrict__ bias,
                                                   float* __restrict__ outF,
                                                   __bf16* __restrict__ outB,
                                                   int M, int Kd, int Nd) {
    int tid  = threadIdx.x;
    int lane = tid & 31;
    int wave = tid >> 5;
    int mtiles = M >> 4;
    int w  = blockIdx.x * 4 + wave;
    int mt = w % mtiles;
    int ng = w / mtiles;
    int colbase = ng * 16 * NT;
    int l = lane & 15, hi = (lane >> 4) & 1;

    v8f acc[NT];
#pragma unroll
    for (int t = 0; t < NT; ++t) acc[t] = vzero();

    for (int k0 = 0; k0 < Kd; k0 += 32) {
        v16bf a = ld_tile(A, Kd, mt * 16, k0, lane);
#pragma unroll
        for (int t = 0; t < NT; ++t) {
            v16bf b = ld_tile(Bt, Kd, colbase + t * 16, k0, lane);
            acc[t] = wmma_bf16(a, b, acc[t]);
        }
    }
#pragma unroll
    for (int t = 0; t < NT; ++t) {
        int col = colbase + t * 16 + l;
        float bv = bias ? bias[col] : 0.f;
#pragma unroll
        for (int i = 0; i < 8; ++i) {
            int row = mt * 16 + i + hi * 8;
            float v = acc[t][i] + bv;
            size_t off = (size_t)row * Nd + col;
            if (outF) outF[off] = v;
            if (outB) outB[off] = f2bf(v);
        }
    }
}

// ---------------------------------------------------------------------------
// Fused per-point attention.  One block per point; wave `w` owns channel slice
// [w*16*NT, (w+1)*16*NT).  A-matrices ([32 x D] bf16, rows 24..31 padding)
// live in a single LDS buffer, reused across the three matmuls.
// ---------------------------------------------------------------------------
template <int D, int NW, int NT>
__device__ inline void attn_mm(const __bf16* sA, const __bf16* Bt, int colbase,
                               int lane, v8f (&acc)[2][NT]) {
    for (int k0 = 0; k0 < D; k0 += 32) {
        v16bf a0 = ld_tile(sA, D, 0, k0, lane);
        v16bf a1 = ld_tile(sA, D, 16, k0, lane);
#pragma unroll
        for (int t = 0; t < NT; ++t) {
            v16bf b = ld_tile(Bt, D, colbase + t * 16, k0, lane);
            acc[0][t] = wmma_bf16(a0, b, acc[0][t]);
            acc[1][t] = wmma_bf16(a1, b, acc[1][t]);
        }
    }
}

template <int D, int NW, int NT>
__global__ __launch_bounds__(NW * 32) void attn_kernel(
    const float* __restrict__ xyz, const int* __restrict__ knn,
    const float* __restrict__ qg, const float* __restrict__ kf,
    const float* __restrict__ vf,
    const float* __restrict__ d1w, const float* __restrict__ d1b,
    const __bf16* __restrict__ d2t, const float* __restrict__ d2b,
    const __bf16* __restrict__ g1t, const float* __restrict__ g1b,
    const __bf16* __restrict__ g2t, const float* __restrict__ g2b,
    __bf16* __restrict__ resb) {
    __shared__ __align__(16) __bf16 sA[32 * D];
    __shared__ int   sIdx[24];
    __shared__ float sNx[24][3];
    __shared__ float sQn[3];

    const int NPT = 4096;
    int pt  = blockIdx.x;
    int bb  = pt >> 12;
    int tid = threadIdx.x;
    int lane = tid & 31, wave = tid >> 5;
    int l = lane & 15, hi = (lane >> 4) & 1;
    int colbase = wave * 16 * NT;

    if (tid < 24) sIdx[tid] = knn[(size_t)pt * 24 + tid];
    __syncthreads();
    if (tid < 24) {
        int nb = sIdx[tid];
        const float* p = xyz + ((size_t)bb * NPT + nb) * 3;
        sNx[tid][0] = p[0]; sNx[tid][1] = p[1]; sNx[tid][2] = p[2];
    }
    if (tid >= 24 && tid < 27) sQn[tid - 24] = xyz[(size_t)pt * 3 + (tid - 24)];
    __syncthreads();

    // pos1 = relu((xyz_n - xyz_j) @ d1_w + d1_b), rows>=24 zero-padded
    for (int e = tid; e < 32 * D; e += NW * 32) {
        int j = e / D, c = e - j * D;
        float v = 0.f;
        if (j < 24) {
            float dx = sQn[0] - sNx[j][0];
            float dy = sQn[1] - sNx[j][1];
            float dz = sQn[2] - sNx[j][2];
            v = dx * d1w[c] + dy * d1w[D + c] + dz * d1w[2 * D + c] + d1b[c];
            v = fmaxf(v, 0.f);
        }
        sA[e] = f2bf(v);
    }
    __syncthreads();

    // pos = pos1 @ d2_w + d2_b   (kept in registers, reused twice later)
    v8f pos[2][NT];
#pragma unroll
    for (int m = 0; m < 2; ++m)
#pragma unroll
        for (int t = 0; t < NT; ++t) pos[m][t] = vzero();
    attn_mm<D, NW, NT>(sA, d2t, colbase, lane, pos);
#pragma unroll
    for (int t = 0; t < NT; ++t) {
        float bv = d2b[colbase + t * 16 + l];
#pragma unroll
        for (int m = 0; m < 2; ++m)
#pragma unroll
            for (int i = 0; i < 8; ++i) pos[m][t][i] += bv;
    }
    __syncthreads();                    // everyone done reading pos1

    // a = q - k_gather + pos  -> sA (bf16)
#pragma unroll
    for (int t = 0; t < NT; ++t) {
        int col = colbase + t * 16 + l;
        float qv = qg[(size_t)pt * D + col];
#pragma unroll
        for (int m = 0; m < 2; ++m)
#pragma unroll
            for (int i = 0; i < 8; ++i) {
                int row = m * 16 + i + hi * 8;
                float av = 0.f;
                if (row < 24) {
                    int nb = sIdx[row];
                    av = qv - kf[((size_t)bb * NPT + nb) * D + col] + pos[m][t][i];
                }
                sA[row * D + col] = f2bf(av);
            }
    }
    __syncthreads();

    // h = relu(a @ g1_w + g1_b)
    v8f hh[2][NT];
#pragma unroll
    for (int m = 0; m < 2; ++m)
#pragma unroll
        for (int t = 0; t < NT; ++t) hh[m][t] = vzero();
    attn_mm<D, NW, NT>(sA, g1t, colbase, lane, hh);
    __syncthreads();                    // everyone done reading a
#pragma unroll
    for (int t = 0; t < NT; ++t) {
        int col = colbase + t * 16 + l;
        float bv = g1b[col];
#pragma unroll
        for (int m = 0; m < 2; ++m)
#pragma unroll
            for (int i = 0; i < 8; ++i) {
                int row = m * 16 + i + hi * 8;
                sA[row * D + col] = f2bf(fmaxf(hh[m][t][i] + bv, 0.f));
            }
    }
    __syncthreads();

    // logits = h @ g2_w + g2_b
    v8f lg[2][NT];
#pragma unroll
    for (int m = 0; m < 2; ++m)
#pragma unroll
        for (int t = 0; t < NT; ++t) lg[m][t] = vzero();
    attn_mm<D, NW, NT>(sA, g2t, colbase, lane, lg);

    // per-channel softmax over the 24 valid rows + weighted sum of (v + pos)
#pragma unroll
    for (int t = 0; t < NT; ++t) {
        int col = colbase + t * 16 + l;
        float bv = g2b[col];
        float mx = -3.4e38f;
#pragma unroll
        for (int m = 0; m < 2; ++m)
#pragma unroll
            for (int i = 0; i < 8; ++i) {
                int row = m * 16 + i + hi * 8;
                if (row < 24) mx = fmaxf(mx, lg[m][t][i] + bv);
            }
        mx = fmaxf(mx, __shfl_xor(mx, 16));
        float sum = 0.f, sres = 0.f;
#pragma unroll
        for (int m = 0; m < 2; ++m)
#pragma unroll
            for (int i = 0; i < 8; ++i) {
                int row = m * 16 + i + hi * 8;
                if (row < 24) {
                    float e = __expf(lg[m][t][i] + bv - mx);
                    sum += e;
                    int nb = sIdx[row];
                    float vv = vf[((size_t)bb * NPT + nb) * D + col] + pos[m][t][i];
                    sres += e * vv;
                }
            }
        sum  += __shfl_xor(sum, 16);
        sres += __shfl_xor(sres, 16);
        float r = sres / sum;
        if (hi == 0) resb[(size_t)pt * D + col] = f2bf(r);
    }
}

// ---------------------------------------------------------------------------
// BatchNorm (training mode): deterministic per-channel block reduction.
// ---------------------------------------------------------------------------
__global__ __launch_bounds__(256) void bn_stats_kernel(const float* __restrict__ y,
                                                       float* __restrict__ mean,
                                                       float* __restrict__ rstd,
                                                       int M, int C) {
    __shared__ float ss[256], sq[256];
    int c = blockIdx.x, tid = threadIdx.x;
    float s = 0.f, q = 0.f;
    for (int r = tid; r < M; r += 256) {
        float v = y[(size_t)r * C + c];
        s += v; q += v * v;
    }
    ss[tid] = s; sq[tid] = q;
    __syncthreads();
    for (int st = 128; st > 0; st >>= 1) {
        if (tid < st) { ss[tid] += ss[tid + st]; sq[tid] += sq[tid + st]; }
        __syncthreads();
    }
    if (tid == 0) {
        float m = ss[0] / (float)M;
        float v = sq[0] / (float)M - m * m;
        mean[c] = m;
        rstd[c] = rsqrtf(v + 1e-5f);
    }
}

__global__ __launch_bounds__(256) void bn_apply_kernel(const float* __restrict__ y,
                                                       const float* __restrict__ mean,
                                                       const float* __restrict__ rstd,
                                                       const float* __restrict__ g,
                                                       const float* __restrict__ bt,
                                                       const float* __restrict__ addsrc,
                                                       float* __restrict__ outF,
                                                       __bf16* __restrict__ outB,
                                                       int M, int C) {
    size_t t = (size_t)blockIdx.x * 256 + threadIdx.x;
    if (t >= (size_t)M * C) return;
    int c = (int)(t % C);
    float v = (y[t] - mean[c]) * rstd[c] * g[c] + bt[c];
    if (addsrc) v += addsrc[t];
    if (outF) outF[t] = v;
    if (outB) outB[t] = f2bf(v);
}

// ---------------------------------------------------------------------------
// Host orchestration
// ---------------------------------------------------------------------------
static void launch_gemm(const __bf16* A, const __bf16* Bt, const float* bias,
                        float* outF, __bf16* outB, int M, int Kd, int Nd,
                        hipStream_t s) {
    if (Nd == 32) {
        int blocks = (M / 16) * (Nd / 32) / 4;
        gemm_kernel<2><<<blocks, 128, 0, s>>>(A, Bt, bias, outF, outB, M, Kd, Nd);
    } else {
        int blocks = (M / 16) * (Nd / 64) / 4;
        gemm_kernel<4><<<blocks, 128, 0, s>>>(A, Bt, bias, outF, outB, M, Kd, Nd);
    }
}

extern "C" void kernel_launch(void* const* d_in, const int* in_sizes, int n_in,
                              void* d_out, int out_size, void* d_ws, size_t ws_size,
                              hipStream_t stream) {
    (void)in_sizes; (void)n_in; (void)out_size; (void)ws_size;
    const int M = 16384;                 // B*N
    const int dims[4] = {32, 64, 128, 512};
    const int tlb[4]  = {1, 18, 35, 52}; // d_in base of each tl block
    auto inf = [&](int i) -> const float* { return (const float*)d_in[i]; };
    const float* xyz = inf(0);

    // -------- workspace carve (~205 MB) --------
    char* w = (char*)d_ws;
    size_t off = 0;
    auto take = [&](size_t bytes) -> void* {
        void* p = w + off;
        off = (off + bytes + 255) & ~(size_t)255;
        return p;
    };
    int*    knn  = (int*)take((size_t)M * 24 * 4);
    __bf16* xb   = (__bf16*)take((size_t)M * 512 * 2);
    float*  qg   = (float*)take((size_t)M * 512 * 4);
    float*  kfb  = (float*)take((size_t)M * 512 * 4);
    float*  vfb  = (float*)take((size_t)M * 512 * 4);
    __bf16* resb = (__bf16*)take((size_t)M * 512 * 2);
    float*  ybuf = (float*)take((size_t)M * 512 * 4);
    __bf16* f1b  = (__bf16*)take((size_t)M * 32 * 2);
    __bf16* f2b  = (__bf16*)take((size_t)M * 64 * 2);
    __bf16* f3b  = (__bf16*)take((size_t)M * 128 * 2);
    float*  rbuf = (float*)take((size_t)M * 512 * 4);
    float*  mean = (float*)take(512 * 4);
    float*  rstd = (float*)take(512 * 4);
    __bf16* wqT  = (__bf16*)take(512 * 512 * 2);
    __bf16* wkT  = (__bf16*)take(512 * 512 * 2);
    __bf16* wvT  = (__bf16*)take(512 * 512 * 2);
    __bf16* d2T  = (__bf16*)take(512 * 512 * 2);
    __bf16* g1T  = (__bf16*)take(512 * 512 * 2);
    __bf16* g2T  = (__bf16*)take(512 * 512 * 2);
    __bf16* fc2T = (__bf16*)take(512 * 512 * 2);
    __bf16* fc1T = (__bf16*)take(512 * 128 * 2);
    __bf16* resT = (__bf16*)take(512 * 64 * 2);

    auto tr = [&](const float* W_, __bf16* Wt_, int Kd, int Nd) {
        transpose_kernel<<<(Kd * Nd + 255) / 256, 256, 0, stream>>>(W_, Wt_, Kd, Nd);
    };

    // KNN
    knn_kernel<<<M / 64, 64, 0, stream>>>(xyz, knn);

    // layer-1 fc1 (Din = 3)
    fc1_l1_kernel<<<(M * 32) / 256, 256, 0, stream>>>(xyz, inf(tlb[0] + 0), inf(tlb[0] + 1), xb, M);

    for (int L = 0; L < 4; ++L) {
        const int D = dims[L];
        const int b = tlb[L];
        // transpose this layer's WMMA weights to bf16 [N][K]
        tr(inf(b + 2),  wqT,  D, D);
        tr(inf(b + 3),  wkT,  D, D);
        tr(inf(b + 4),  wvT,  D, D);
        tr(inf(b + 7),  d2T,  D, D);
        tr(inf(b + 9),  g1T,  D, D);
        tr(inf(b + 11), g2T,  D, D);
        tr(inf(b + 13), fc2T, D, D);

        // q / k / v projections
        launch_gemm(xb, wqT, nullptr, qg,  nullptr, M, D, D, stream);
        launch_gemm(xb, wkT, nullptr, kfb, nullptr, M, D, D, stream);
        launch_gemm(xb, wvT, nullptr, vfb, nullptr, M, D, D, stream);

        // fused attention
        const float* d1w = inf(b + 5);
        const float* d1b = inf(b + 6);
        const float* d2b = inf(b + 8);
        const float* g1b = inf(b + 10);
        const float* g2b = inf(b + 12);
        switch (L) {
        case 0: attn_kernel<32, 2, 1><<<M, 64, 0, stream>>>(xyz, knn, qg, kfb, vfb, d1w, d1b, d2T, d2b, g1T, g1b, g2T, g2b, resb); break;
        case 1: attn_kernel<64, 4, 1><<<M, 128, 0, stream>>>(xyz, knn, qg, kfb, vfb, d1w, d1b, d2T, d2b, g1T, g1b, g2T, g2b, resb); break;
        case 2: attn_kernel<128, 8, 1><<<M, 256, 0, stream>>>(xyz, knn, qg, kfb, vfb, d1w, d1b, d2T, d2b, g1T, g1b, g2T, g2b, resb); break;
        default: attn_kernel<512, 8, 4><<<M, 256, 0, stream>>>(xyz, knn, qg, kfb, vfb, d1w, d1b, d2T, d2b, g1T, g1b, g2T, g2b, resb); break;
        }

        // fc2
        launch_gemm(resb, fc2T, inf(b + 14), ybuf, nullptr, M, D, D, stream);

        // BatchNorm
        bn_stats_kernel<<<D, 256, 0, stream>>>(ybuf, mean, rstd, M, D);
        if (L < 3) {
            __bf16* outB = (L == 0) ? f1b : (L == 1) ? f2b : f3b;
            bn_apply_kernel<<<(M * D) / 256, 256, 0, stream>>>(
                ybuf, mean, rstd, inf(b + 15), inf(b + 16), nullptr, nullptr, outB, M, D);
            if (L == 1) { // residual branch: r = f2 @ res_w + res_b
                tr(inf(69), resT, 64, 512);
                launch_gemm(f2b, resT, inf(70), rbuf, nullptr, M, 64, 512, stream);
            }
            // next layer's fc1
            int Dn = dims[L + 1];
            tr(inf(tlb[L + 1] + 0), fc1T, D, Dn);
            launch_gemm(outB, fc1T, inf(tlb[L + 1] + 1), nullptr, xb, M, D, Dn, stream);
        } else {
            // f4 = BN4(...) + residual, straight to d_out (f32)
            bn_apply_kernel<<<(M * D) / 256, 256, 0, stream>>>(
                ybuf, mean, rstd, inf(b + 15), inf(b + 16), rbuf, (float*)d_out, nullptr, M, D);
        }
    }
}
